// Locate_Subject_89610197664175
// MI455X (gfx1250) — matrix operand order repaired and checked by reading, not compiled, not run on hardware
//
#include <hip/hip_runtime.h>
#include <hip/hip_bf16.h>

typedef __attribute__((ext_vector_type(16))) __bf16 v16bf;
typedef __attribute__((ext_vector_type(8)))  float  v8f;
typedef __attribute__((ext_vector_type(16))) float  v16f;

#define B_SZ 64
#define T_SZ 1024
#define H_SZ 256
#define K_SZ 4

// fp32 -> bf16 round-to-nearest-even (host/packing path)
__device__ __host__ inline __bf16 f2bf(float f) {
    union { float f; unsigned u; } in; in.f = f;
    unsigned r = in.u + 0x7FFFu + ((in.u >> 16) & 1u);
    union { unsigned short s; __bf16 b; } out; out.s = (unsigned short)(r >> 16);
    return out.b;
}

// ---------------------------------------------------------------------------
// Pack conv_w [H_out][H_in][3] fp32 into per-fragment bf16 B-matrix layout:
// wp[(((tap*8+kc)*16 + no)*32 + lane)*16 + j]
//   = W[ho = no*16 + lane%16][hi = kc*32 + (j<8 ? 8h+j : 16+8h+(j-8)), h = lane/16][tap]
// (B fragment 32x16: lanes 0-15 hold K=0..7,16..23 of col n=lane, lanes 16-31 K=8..15,24..31)
// ---------------------------------------------------------------------------
__global__ void pack_weights_kernel(const float* __restrict__ cw,
                                    __bf16* __restrict__ wp) {
    int idx = blockIdx.x * blockDim.x + threadIdx.x;
    if (idx >= 3 * 8 * 16 * 32 * 16) return;
    int j   =  idx        & 15;
    int l   = (idx >> 4)  & 31;
    int no  = (idx >> 9)  & 15;
    int kc  = (idx >> 13) & 7;
    int tap =  idx >> 16;
    int ho  = no * 16 + (l & 15);
    int hi  = kc * 32 + ((j & 8) << 1) + ((l < 16) ? 0 : 8) + (j & 7);
    wp[idx] = f2bf(cw[(ho * H_SZ + hi) * 3 + tap]);
}

// ---------------------------------------------------------------------------
// Fused conv1d(k=3) + bias + ReLU + linear(256->4) via bf16 WMMA.
// One wave per 16-timestep tile. 384 v_wmma_f32_16x16x32_bf16 per wave.
// ---------------------------------------------------------------------------
__global__ __launch_bounds__(128) void emissions_wmma_kernel(
    const float* __restrict__ x,      // [B,T,H]
    const float* __restrict__ cb,     // [H]
    const float* __restrict__ lw,     // [K,H]
    const float* __restrict__ lb,     // [K]
    const __bf16* __restrict__ wp,    // packed weights
    float* __restrict__ em)           // [B,T,K]
{
    const int lid  = threadIdx.x & 31;
    const int wv   = threadIdx.x >> 5;
    const int tile = blockIdx.x * 4 + wv;          // 4096 tiles total
    const int b    = tile >> 6;                    // 64 tiles per batch row
    const int t0   = (tile & 63) << 4;
    const int m    = lid & 15;                     // row within 16-row tile
    const int hiSel = (lid < 16) ? 0 : 8;          // A-fragment K-half select

    // Per-tap clamped row pointers + validity (conv zero padding at t edges)
    const float* xr[3];
    bool valid[3];
#pragma unroll
    for (int tap = 0; tap < 3; ++tap) {
        int t  = t0 + m + tap - 1;
        int tc = t < 0 ? 0 : (t > T_SZ - 1 ? T_SZ - 1 : t);
        xr[tap]    = x + ((size_t)b * T_SZ + tc) * H_SZ;
        valid[tap] = (t >= 0 && t < T_SZ);
    }

    const v16bf bfzero = {};

    float emAcc[32];                               // [8 rows][4 tags] partials
#pragma unroll
    for (int i = 0; i < 32; ++i) emAcc[i] = 0.0f;

#pragma unroll 1
    for (int half = 0; half < 2; ++half) {
        v8f c[8];
#pragma unroll
        for (int j = 0; j < 8; ++j) c[j] = (v8f){0, 0, 0, 0, 0, 0, 0, 0};

#pragma unroll 1
        for (int tap = 0; tap < 3; ++tap) {
            const bool ok = valid[tap];
            const float* p = xr[tap];
#pragma unroll 1                                   // keep rolled: avoid LICM hoisting 24 A-frags
            for (int kc = 0; kc < 8; ++kc) {
                const int coff = kc * 32 + hiSel;
                // K-low group (8 floats) and K-high group (8 floats)
                const v8f qlo = *(const v8f*)(p + coff);
                const v8f qhi = *(const v8f*)(p + coff + 16);
                const v16f qf = __builtin_shufflevector(
                    qlo, qhi, 0, 1, 2, 3, 4, 5, 6, 7,
                              8, 9, 10, 11, 12, 13, 14, 15);
                v16bf a = __builtin_convertvector(qf, v16bf);  // hw pack-cvt
                if (!ok) a = bfzero;                           // zero padding

                const size_t fbase = (size_t)((tap * 8 + kc) * 16 + half * 8) * 512;
#pragma unroll
                for (int j = 0; j < 8; ++j) {
                    const v16bf bf =
                        *(const v16bf*)(wp + fbase + (size_t)j * 512 + lid * 16);
                    c[j] = __builtin_amdgcn_wmma_f32_16x16x32_bf16(
                        false, a, false, bf, (short)0, c[j], false, false);
                }
            }
        }

        // Epilogue: bias + ReLU + fuse 256->4 linear into per-lane partials
#pragma unroll
        for (int j = 0; j < 8; ++j) {
            const int ho = (half * 8 + j) * 16 + (lid & 15);
            const float bias = cb[ho];
            const float l0 = lw[0 * H_SZ + ho];
            const float l1 = lw[1 * H_SZ + ho];
            const float l2 = lw[2 * H_SZ + ho];
            const float l3 = lw[3 * H_SZ + ho];
#pragma unroll
            for (int r = 0; r < 8; ++r) {
                float y = c[j][r] + bias;
                y = y > 0.0f ? y : 0.0f;
                emAcc[r * 4 + 0] += y * l0;
                emAcc[r * 4 + 1] += y * l1;
                emAcc[r * 4 + 2] += y * l2;
                emAcc[r * 4 + 3] += y * l3;
            }
        }
    }

    // Reduce partials across the 16 lanes of each half-group
#pragma unroll
    for (int i = 0; i < 32; ++i) {
        emAcc[i] += __shfl_xor(emAcc[i], 1, 32);
        emAcc[i] += __shfl_xor(emAcc[i], 2, 32);
        emAcc[i] += __shfl_xor(emAcc[i], 4, 32);
        emAcc[i] += __shfl_xor(emAcc[i], 8, 32);
    }

    if (lid == 0 || lid == 16) {
        const int mb = (lid == 0) ? 0 : 8;
        const float b0 = lb[0], b1 = lb[1], b2 = lb[2], b3 = lb[3];
#pragma unroll
        for (int r = 0; r < 8; ++r) {
            float4 o;
            o.x = emAcc[r * 4 + 0] + b0;
            o.y = emAcc[r * 4 + 1] + b1;
            o.z = emAcc[r * 4 + 2] + b2;
            o.w = emAcc[r * 4 + 3] + b3;
            *(float4*)(em + ((size_t)b * T_SZ + t0 + mb + r) * K_SZ) = o;
        }
    }
}

// ---------------------------------------------------------------------------
// Viterbi decode, K=4. One thread per batch row; history packed 4 ptrs/u32.
// ---------------------------------------------------------------------------
__global__ void viterbi_kernel(const float* __restrict__ em,
                               const float* __restrict__ st,
                               const float* __restrict__ en,
                               const float* __restrict__ tr,
                               unsigned int* __restrict__ hist,  // [T-1][B]
                               int* __restrict__ path)           // [B,T]
{
    const int b = blockIdx.x * blockDim.x + threadIdx.x;
    if (b >= B_SZ) return;

    float trr[16];
#pragma unroll
    for (int i = 0; i < 16; ++i) trr[i] = tr[i];

    const float* eb = em + (size_t)b * T_SZ * K_SZ;
    float s[4];
#pragma unroll
    for (int k = 0; k < 4; ++k) s[k] = st[k] + eb[k];

    for (int t = 1; t < T_SZ; ++t) {
        const float4 ev = *(const float4*)(eb + t * 4);
        const float e4[4] = {ev.x, ev.y, ev.z, ev.w};
        float ns[4];
        unsigned h = 0;
#pragma unroll
        for (int kn = 0; kn < 4; ++kn) {
            float best = s[0] + trr[0 * 4 + kn];
            int bi = 0;
#pragma unroll
            for (int kp = 1; kp < 4; ++kp) {
                float v = s[kp] + trr[kp * 4 + kn];
                if (v > best) { best = v; bi = kp; }   // '>' keeps first max (jnp.argmax)
            }
            ns[kn] = best + e4[kn];
            h |= (unsigned)bi << (8 * kn);
        }
        hist[(size_t)(t - 1) * B_SZ + b] = h;
#pragma unroll
        for (int k = 0; k < 4; ++k) s[k] = ns[k];
    }

#pragma unroll
    for (int k = 0; k < 4; ++k) s[k] += en[k];
    int last = 0;
    float best = s[0];
#pragma unroll
    for (int k = 1; k < 4; ++k)
        if (s[k] > best) { best = s[k]; last = k; }

    path[(size_t)b * T_SZ + T_SZ - 1] = last;
    int cur = last;
    for (int t = T_SZ - 2; t >= 0; --t) {
        unsigned h = hist[(size_t)t * B_SZ + b];   // address independent of cur -> pipelined
        cur = (int)((h >> (8 * cur)) & 255u);
        path[(size_t)b * T_SZ + t] = cur;
    }
}

// ---------------------------------------------------------------------------
extern "C" void kernel_launch(void* const* d_in, const int* in_sizes, int n_in,
                              void* d_out, int out_size, void* d_ws, size_t ws_size,
                              hipStream_t stream) {
    (void)in_sizes; (void)n_in; (void)out_size; (void)ws_size;

    const float* x      = (const float*)d_in[0];  // [B,T,H]
    const float* conv_w = (const float*)d_in[1];  // [H,H,3]
    const float* conv_b = (const float*)d_in[2];  // [H]
    const float* lin_w  = (const float*)d_in[3];  // [K,H]
    const float* lin_b  = (const float*)d_in[4];  // [K]
    const float* crf_s  = (const float*)d_in[5];  // [K]
    const float* crf_e  = (const float*)d_in[6];  // [K]
    const float* crf_t  = (const float*)d_in[7];  // [K,K]

    char* ws = (char*)d_ws;
    __bf16*       wp   = (__bf16*)ws;                            // 393,216 B
    float*        em   = (float*)(ws + 393216);                  // 1,048,576 B
    unsigned int* hist = (unsigned int*)(ws + 393216 + 1048576); // 261,888 B
    int*          path = (int*)d_out;                            // [B,T] int32

    // 1) pack conv weights into WMMA B-fragment layout (bf16)
    pack_weights_kernel<<<768, 256, 0, stream>>>(conv_w, wp);

    // 2) fused conv+relu+linear emissions via bf16 WMMA
    emissions_wmma_kernel<<<1024, 128, 0, stream>>>(x, conv_b, lin_w, lin_b, wp, em);

    // 3) Viterbi decode
    viterbi_kernel<<<2, 32, 0, stream>>>(em, crf_s, crf_e, crf_t, hist, path);
}
